// SlideXL_26955214749734
// MI455X (gfx1250) — matrix operand order repaired and checked by reading, not compiled, not run on hardware
//
#include <hip/hip_runtime.h>
#include <math.h>

// ---------------- CDNA5 (gfx1250) types ----------------
typedef _Float16 F16;
typedef _Float16 v16h __attribute__((ext_vector_type(16)));
typedef _Float16 v8h  __attribute__((ext_vector_type(8)));
typedef float    v8f  __attribute__((ext_vector_type(8)));

// Model constants (from reference)
#define D_V 1024
#define D_T 768
#define DM  768
#define DFF 3072
#define NL  4
#define NH  12
#define DH  64
#define SSTR 32
#define CHUNK 1024
#define GROUPS 32              // CHUNK / SSTR
#define HL 1056                // GROUPS * (SSTR+1)
#define NPATCH 8192
#define NCHUNKS 8
#define NREP 128
#define MEMMAX 256

// ---------------- f32 -> f16 convert ----------------
__global__ __launch_bounds__(256) void k_cvt(const float* __restrict__ s,
                                             F16* __restrict__ d, int n) {
  int i = blockIdx.x * 256 + threadIdx.x;
  if (i < n) d[i] = (F16)s[i];
}

// ---------------- generic WMMA GEMM ----------------
// C[M,N] = epilogue( scale * A[M,K] @ B[K,N] + bias, gelu, +resid )
// A,B f16 row-major; C f32 (optional); Cb f16 copy (optional); batched via z.
struct GemmArgs {
  const F16* A; long lda, sA;
  const F16* B; long ldb, sB;
  float* C; long ldc, sC;
  F16* Cb; long ldcb, sCb;
  const float* bias;
  const float* resid; long ldr;
  int M, N, K;
  float scale;
  int gelu;
};

__global__ __launch_bounds__(128) void k_gemm(GemmArgs g) {
  __shared__ alignas(32) F16 Bs[32 * 64];
  const int wave = threadIdx.x >> 5;
  const int lane = threadIdx.x & 31;
  const long bi = blockIdx.z;
  const F16* A = g.A + bi * g.sA;
  const F16* B = g.B + bi * g.sB;
  const int n0 = blockIdx.x * 64;
  int m0 = blockIdx.y * 64 + wave * 16;
  if (m0 + 16 > g.M) m0 = g.M - 16;   // M is always a multiple of 16; dup work is benign
  const int kb = (lane < 16) ? 0 : 8; // ISA 16-bit A-fragment layout
  const F16* Arow = A + (long)(m0 + (lane & 15)) * g.lda + kb;

  v8f acc0 = {}, acc1 = {}, acc2 = {}, acc3 = {};
  for (int k0 = 0; k0 < g.K; k0 += 32) {
    // stage B[k0..k0+32)[n0..n0+64) into LDS (128 threads x 16 halves)
    const int r  = threadIdx.x >> 2;
    const int cc = (threadIdx.x & 3) * 16;
    *(v16h*)(&Bs[r * 64 + cc]) = *(const v16h*)(B + (long)(k0 + r) * g.ldb + n0 + cc);
    __syncthreads();
    // A fragment: halves 0-7 = K kb..kb+7, halves 8-15 = K kb+16..kb+23
    v8h a0 = *(const v8h*)(Arow + k0);
    v8h a1 = *(const v8h*)(Arow + k0 + 16);
    v16h af = __builtin_shufflevector(a0, a1, 0,1,2,3,4,5,6,7,8,9,10,11,12,13,14,15);
    // B fragment: lane holds N=lane&15, halves hold K in the same order as A
    const int bn = lane & 15;
    v16h b0, b1, b2, b3;
    for (int s = 0; s < 16; ++s) {
      const int kk = kb + (s < 8 ? s : s + 8);
      const F16* brow = &Bs[kk * 64 + bn];
      b0[s] = brow[0]; b1[s] = brow[16]; b2[s] = brow[32]; b3[s] = brow[48];
    }
    acc0 = __builtin_amdgcn_wmma_f32_16x16x32_f16(false, af, false, b0, (short)0, acc0, false, false);
    acc1 = __builtin_amdgcn_wmma_f32_16x16x32_f16(false, af, false, b1, (short)0, acc1, false, false);
    acc2 = __builtin_amdgcn_wmma_f32_16x16x32_f16(false, af, false, b2, (short)0, acc2, false, false);
    acc3 = __builtin_amdgcn_wmma_f32_16x16x32_f16(false, af, false, b3, (short)0, acc3, false, false);
    __syncthreads();
  }
  // epilogue: C layout = lane&15 -> col, VGPR r -> row m0 + r + (lane>=16 ? 8 : 0)
  const int colL  = lane & 15;
  const int rbase = m0 + ((lane < 16) ? 0 : 8);
  v8f accs[4] = {acc0, acc1, acc2, acc3};
#pragma unroll
  for (int t = 0; t < 4; ++t) {
    const int col = n0 + t * 16 + colL;
    const float bb = g.bias ? g.bias[col] : 0.f;
#pragma unroll
    for (int rr = 0; rr < 8; ++rr) {
      const long row = rbase + rr;
      float v = accs[t][rr] * g.scale + bb;
      if (g.gelu) v = 0.5f * v * (1.f + erff(v * 0.70710678118654752f));
      if (g.resid) v += g.resid[row * g.ldr + col];
      if (g.C)  g.C [bi * g.sC  + row * g.ldc  + col] = v;
      if (g.Cb) g.Cb[bi * g.sCb + row * g.ldcb + col] = (F16)v;
    }
  }
}

// ---------------- RMS norm (f32 in, f16 out) ----------------
__global__ __launch_bounds__(256) void k_rms(const float* __restrict__ x,
                                             const float* __restrict__ gain,
                                             F16* __restrict__ out, int ncols) {
  __shared__ float red[256];
  const long row = blockIdx.x;
  const float* xr = x + row * ncols;
  float s = 0.f;
  for (int i = threadIdx.x; i < ncols; i += 256) { float v = xr[i]; s += v * v; }
  red[threadIdx.x] = s; __syncthreads();
  for (int o = 128; o > 0; o >>= 1) { if (threadIdx.x < o) red[threadIdx.x] += red[threadIdx.x + o]; __syncthreads(); }
  const float r = rsqrtf(red[0] / (float)ncols + 1e-6f);
  F16* op = out + row * ncols;
  for (int i = threadIdx.x; i < ncols; i += 256) op[i] = (F16)(xr[i] * r * gain[i]);
}

// ---------------- softmax in place over f16 score rows ----------------
__global__ __launch_bounds__(256) void k_softmax(F16* __restrict__ s, int rowsPerHead,
                                                 int L, int Lpad) {
  __shared__ float red[256];
  const long row = (long)blockIdx.y * rowsPerHead + blockIdx.x;
  F16* p = s + row * (long)Lpad;
  float m = -1e30f;
  for (int i = threadIdx.x; i < L; i += 256) m = fmaxf(m, (float)p[i]);
  red[threadIdx.x] = m; __syncthreads();
  for (int o = 128; o > 0; o >>= 1) { if (threadIdx.x < o) red[threadIdx.x] = fmaxf(red[threadIdx.x], red[threadIdx.x + o]); __syncthreads(); }
  m = red[0]; __syncthreads();
  float sum = 0.f;
  for (int i = threadIdx.x; i < L; i += 256) sum += expf((float)p[i] - m);
  red[threadIdx.x] = sum; __syncthreads();
  for (int o = 128; o > 0; o >>= 1) { if (threadIdx.x < o) red[threadIdx.x] += red[threadIdx.x + o]; __syncthreads(); }
  const float inv = 1.f / red[0];
  for (int i = threadIdx.x; i < L; i += 256) p[i] = (F16)(expf((float)p[i] - m) * inv);
  for (int i = L + threadIdx.x; i < Lpad; i += 256) p[i] = (F16)0.f;
}

// ---------------- kc^T builder: out[h][d][j] (zero padded to Lpad) ----------------
__global__ __launch_bounds__(256) void k_build_kT(const F16* __restrict__ cur, int curLen,
                                                  const F16* __restrict__ mem, int memLen,
                                                  F16* __restrict__ out, int Lpad) {
  long idx = (long)blockIdx.x * 256 + threadIdx.x;
  const long total = (long)NH * DH * Lpad;
  if (idx >= total) return;
  int j = (int)(idx % Lpad); long t = idx / Lpad;
  int d = (int)(t % DH); int h = (int)(t / DH);
  F16 v = (F16)0.f;
  if (j < memLen) v = mem[((long)h * MEMMAX + j) * DH + d];
  else if (j - memLen < curLen) v = cur[(long)(j - memLen) * DM + h * DH + d];
  out[idx] = v;
}

// ---------------- vc builder: out[h][j][d] (zero padded) ----------------
__global__ __launch_bounds__(256) void k_build_vc(const F16* __restrict__ cur, int curLen,
                                                  const F16* __restrict__ mem, int memLen,
                                                  F16* __restrict__ out, int Lpad) {
  long idx = (long)blockIdx.x * 256 + threadIdx.x;
  const long total = (long)NH * Lpad * DH;
  if (idx >= total) return;
  int d = (int)(idx % DH); long t = idx / DH;
  int j = (int)(t % Lpad); int h = (int)(t / Lpad);
  F16 v = (F16)0.f;
  if (j < memLen) v = mem[((long)h * MEMMAX + j) * DH + d];
  else if (j - memLen < curLen) v = cur[(long)(j - memLen) * DM + h * DH + d];
  out[idx] = v;
}

// ---------------- append k,v at SST positions into per-layer memory ----------------
__global__ __launch_bounds__(256) void k_update_mem(const F16* __restrict__ k16,
                                                    const F16* __restrict__ v16,
                                                    F16* __restrict__ memk,
                                                    F16* __restrict__ memv, int cOff) {
  int idx = blockIdx.x * 256 + threadIdx.x;
  if (idx >= GROUPS * DM) return;
  int grp = idx / DM, col = idx % DM;
  int h = col / DH, d = col % DH;
  long src = (long)(grp * (SSTR + 1) + SSTR) * DM + col;
  long dst = ((long)h * MEMMAX + cOff + grp) * DH + d;
  memk[dst] = k16[src];
  memv[dst] = v16[src];
}

// ---------------- assemble h = [32 tokens | sst] x 32 groups ----------------
__global__ __launch_bounds__(256) void k_build_h(const float* __restrict__ x,
                                                 const float* __restrict__ sst,
                                                 float* __restrict__ h) {
  int row = blockIdx.x;                 // 0..HL-1
  int g = row / (SSTR + 1), r = row % (SSTR + 1);
  const float* src = (r < SSTR) ? (x + (long)(g * SSTR + r) * DM) : sst;
  float* dst = h + (long)row * DM;
  for (int i = threadIdx.x; i < DM; i += 256) dst[i] = src[i];
}

// ---------------- collect SST rows of h into slide memory (f16) ----------------
__global__ __launch_bounds__(256) void k_slide(const float* __restrict__ h,
                                               F16* __restrict__ smem16, int cOff) {
  int idx = blockIdx.x * 256 + threadIdx.x;
  if (idx >= GROUPS * DM) return;
  int g = idx / DM, col = idx % DM;
  float v = h[(long)(g * (SSTR + 1) + SSTR) * DM + col];
  smem16[(long)(cOff + g) * DM + col] = (F16)v;
}

// ---------------- mean pool + classifier head ----------------
__global__ __launch_bounds__(768) void k_head(const float* __restrict__ q,
                                              const float* __restrict__ hw,
                                              const float* __restrict__ hb,
                                              float* __restrict__ out) {
  __shared__ float pooled[DM];
  int d = threadIdx.x;
  float s = 0.f;
  for (int i = 0; i < NREP; ++i) s += q[(long)i * DM + d];
  pooled[d] = s * (1.f / (float)NREP);
  __syncthreads();
  if (d < 2) {
    float acc = hb[d];
    for (int kk = 0; kk < DM; ++kk) acc += pooled[kk] * hw[kk * 2 + d];
    out[d] = acc;
  }
}

// ---------------- host-side GEMM launcher ----------------
static inline void launch_gemm(hipStream_t st,
                               const F16* A, long lda, long sA,
                               const F16* B, long ldb, long sB,
                               float* C, long ldc, long sC,
                               F16* Cb, long ldcb, long sCb,
                               const float* bias, const float* resid, long ldr,
                               int M, int N, int K, float scale, int gelu, int batch) {
  GemmArgs g;
  g.A = A; g.lda = lda; g.sA = sA;
  g.B = B; g.ldb = ldb; g.sB = sB;
  g.C = C; g.ldc = ldc; g.sC = sC;
  g.Cb = Cb; g.ldcb = ldcb; g.sCb = sCb;
  g.bias = bias; g.resid = resid; g.ldr = ldr;
  g.M = M; g.N = N; g.K = K; g.scale = scale; g.gelu = gelu;
  dim3 grid((unsigned)(N / 64), (unsigned)((M + 63) / 64), (unsigned)batch);
  k_gemm<<<grid, dim3(128), 0, st>>>(g);
}

extern "C" void kernel_launch(void* const* d_in, const int* in_sizes, int n_in,
                              void* d_out, int out_size, void* d_ws, size_t ws_size,
                              hipStream_t stream) {
  (void)in_sizes; (void)n_in; (void)out_size; (void)ws_size;
  // Input order: patch_feats, report_feats, then params leaves in jax tree order
  // (dict keys sorted at every level), then chunk_size.
  const float* patch  = (const float*)d_in[0];
  const float* report = (const float*)d_in[1];
  const float* cross_wk = (const float*)d_in[2];
  const float* cross_wo = (const float*)d_in[3];
  const float* cross_wq = (const float*)d_in[4];
  const float* cross_wv = (const float*)d_in[5];
  const float* head_b   = (const float*)d_in[6];
  const float* head_w   = (const float*)d_in[7];
  // layers[l]: 8+8l + {0:n1,1:n2,2:w1,3:w2,4:wk,5:wo,6:wq,7:wv}
  const float* proj_b1 = (const float*)d_in[40];
  const float* proj_b2 = (const float*)d_in[41];
  const float* proj_w1 = (const float*)d_in[42];
  const float* proj_w2 = (const float*)d_in[43];
  const float* rff_w1  = (const float*)d_in[44];
  const float* rff_w2  = (const float*)d_in[45];
  const float* rn1     = (const float*)d_in[46];
  const float* rn2     = (const float*)d_in[47];
  const float* sst     = (const float*)d_in[48];
  const float* tp_b1   = (const float*)d_in[49];
  const float* tp_b2   = (const float*)d_in[50];
  const float* tp_w1   = (const float*)d_in[51];
  const float* tp_w2   = (const float*)d_in[52];

  // ---- workspace carving (256B aligned) ----
  char* base = (char*)d_ws;
  size_t off = 0;
  auto alloc = [&](size_t bytes) -> void* {
    off = (off + 255) & ~(size_t)255;
    void* p = base + off;
    off += bytes;
    return p;
  };
  auto cvt = [&](const float* s, F16* d, long n) {
    k_cvt<<<dim3((unsigned)((n + 255) / 256)), dim3(256), 0, stream>>>(s, d, (int)n);
  };

  // f16 weights
  F16* Wck = (F16*)alloc((size_t)DM * DM * 2);
  F16* Wco = (F16*)alloc((size_t)DM * DM * 2);
  F16* Wcq = (F16*)alloc((size_t)DM * DM * 2);
  F16* Wcv = (F16*)alloc((size_t)DM * DM * 2);
  F16* Wp1 = (F16*)alloc((size_t)D_V * DM * 2);
  F16* Wp2 = (F16*)alloc((size_t)DM * DM * 2);
  F16* Wt1 = (F16*)alloc((size_t)D_T * DM * 2);
  F16* Wt2 = (F16*)alloc((size_t)DM * DM * 2);
  F16* Wrf1 = (F16*)alloc((size_t)DM * DFF * 2);
  F16* Wrf2 = (F16*)alloc((size_t)DFF * DM * 2);
  F16 *Wq16[NL], *Wk16[NL], *Wv16[NL], *Wo16[NL], *W116[NL], *W216[NL];
  const float *n1g[NL], *n2g[NL];
  for (int l = 0; l < NL; ++l) {
    n1g[l] = (const float*)d_in[8 + 8 * l + 0];
    n2g[l] = (const float*)d_in[8 + 8 * l + 1];
    W116[l] = (F16*)alloc((size_t)DM * DFF * 2);
    W216[l] = (F16*)alloc((size_t)DFF * DM * 2);
    Wk16[l] = (F16*)alloc((size_t)DM * DM * 2);
    Wo16[l] = (F16*)alloc((size_t)DM * DM * 2);
    Wq16[l] = (F16*)alloc((size_t)DM * DM * 2);
    Wv16[l] = (F16*)alloc((size_t)DM * DM * 2);
  }

  // activations
  const int LPADMAX = 1280;  // max padded kv length
  F16*  P16   = (F16*)alloc((size_t)NPATCH * D_V * 2);
  F16*  rep16 = (F16*)alloc((size_t)NREP * D_T * 2);
  F16*  t1_16 = (F16*)alloc((size_t)NPATCH * DM * 2);
  float* x32  = (float*)alloc((size_t)NPATCH * DM * 4);
  float* h32  = (float*)alloc((size_t)HL * DM * 4);
  F16*  hn16  = (F16*)alloc((size_t)HL * DM * 2);
  F16*  q16   = (F16*)alloc((size_t)HL * DM * 2);
  F16*  k16   = (F16*)alloc((size_t)HL * DM * 2);
  F16*  v16   = (F16*)alloc((size_t)HL * DM * 2);
  F16*  y16   = (F16*)alloc((size_t)HL * DM * 2);
  F16*  ff16  = (F16*)alloc((size_t)HL * DFF * 2);
  F16*  kT16  = (F16*)alloc((size_t)NH * DH * LPADMAX * 2);
  F16*  vc16  = (F16*)alloc((size_t)NH * LPADMAX * DH * 2);
  F16*  s16   = (F16*)alloc((size_t)NH * HL * LPADMAX * 2);
  F16*  memk  = (F16*)alloc((size_t)NL * NH * MEMMAX * DH * 2);
  F16*  memv  = (F16*)alloc((size_t)NL * NH * MEMMAX * DH * 2);
  F16*  smem16 = (F16*)alloc((size_t)MEMMAX * DM * 2);
  float* rq32 = (float*)alloc((size_t)NREP * DM * 4);
  F16*  rqn16 = (F16*)alloc((size_t)NREP * DM * 2);
  F16*  t2_16 = (F16*)alloc((size_t)NREP * DM * 2);
  F16*  cq16  = (F16*)alloc((size_t)NREP * DM * 2);
  F16*  ck16  = (F16*)alloc((size_t)MEMMAX * DM * 2);
  F16*  cv16  = (F16*)alloc((size_t)MEMMAX * DM * 2);
  F16*  cy16  = (F16*)alloc((size_t)NREP * DM * 2);
  F16*  cf1   = (F16*)alloc((size_t)NREP * DFF * 2);

  // ---- weight / input conversions ----
  cvt(cross_wk, Wck, (long)DM * DM);
  cvt(cross_wo, Wco, (long)DM * DM);
  cvt(cross_wq, Wcq, (long)DM * DM);
  cvt(cross_wv, Wcv, (long)DM * DM);
  cvt(proj_w1, Wp1, (long)D_V * DM);
  cvt(proj_w2, Wp2, (long)DM * DM);
  cvt(tp_w1, Wt1, (long)D_T * DM);
  cvt(tp_w2, Wt2, (long)DM * DM);
  cvt(rff_w1, Wrf1, (long)DM * DFF);
  cvt(rff_w2, Wrf2, (long)DFF * DM);
  for (int l = 0; l < NL; ++l) {
    cvt((const float*)d_in[8 + 8 * l + 2], W116[l], (long)DM * DFF);
    cvt((const float*)d_in[8 + 8 * l + 3], W216[l], (long)DFF * DM);
    cvt((const float*)d_in[8 + 8 * l + 4], Wk16[l], (long)DM * DM);
    cvt((const float*)d_in[8 + 8 * l + 5], Wo16[l], (long)DM * DM);
    cvt((const float*)d_in[8 + 8 * l + 6], Wq16[l], (long)DM * DM);
    cvt((const float*)d_in[8 + 8 * l + 7], Wv16[l], (long)DM * DM);
  }
  cvt(patch, P16, (long)NPATCH * D_V);
  cvt(report, rep16, (long)NREP * D_T);

  // ---- patch MLP projection: x = gelu(patch@W1+b1)@W2+b2 ----
  launch_gemm(stream, P16, D_V, 0, Wp1, DM, 0, nullptr, 0, 0, t1_16, DM, 0,
              proj_b1, nullptr, 0, NPATCH, DM, D_V, 1.f, 1, 1);
  launch_gemm(stream, t1_16, DM, 0, Wp2, DM, 0, x32, DM, 0, nullptr, 0, 0,
              proj_b2, nullptr, 0, NPATCH, DM, DM, 1.f, 0, 1);

  // ---- chunked transformer with SST KV memory ----
  for (int c = 0; c < NCHUNKS; ++c) {
    k_build_h<<<dim3(HL), dim3(256), 0, stream>>>(x32 + (long)c * CHUNK * DM, sst, h32);
    const int memLen = SSTR * c;           // <= 224; never exceeds MEMMAX here
    const int L = memLen + HL;
    const int Lpad = (L + 63) & ~63;
    for (int l = 0; l < NL; ++l) {
      F16* mk = memk + (long)l * NH * MEMMAX * DH;
      F16* mv = memv + (long)l * NH * MEMMAX * DH;
      // pre-norm + QKV
      k_rms<<<dim3(HL), dim3(256), 0, stream>>>(h32, n1g[l], hn16, DM);
      launch_gemm(stream, hn16, DM, 0, Wq16[l], DM, 0, nullptr, 0, 0, q16, DM, 0,
                  nullptr, nullptr, 0, HL, DM, DM, 1.f, 0, 1);
      launch_gemm(stream, hn16, DM, 0, Wk16[l], DM, 0, nullptr, 0, 0, k16, DM, 0,
                  nullptr, nullptr, 0, HL, DM, DM, 1.f, 0, 1);
      launch_gemm(stream, hn16, DM, 0, Wv16[l], DM, 0, nullptr, 0, 0, v16, DM, 0,
                  nullptr, nullptr, 0, HL, DM, DM, 1.f, 0, 1);
      // kc^T / vc with memory prefix + zero padding
      {
        long tot = (long)NH * DH * Lpad;
        k_build_kT<<<dim3((unsigned)((tot + 255) / 256)), dim3(256), 0, stream>>>(
            k16, HL, mk, memLen, kT16, Lpad);
        k_build_vc<<<dim3((unsigned)((tot + 255) / 256)), dim3(256), 0, stream>>>(
            v16, HL, mv, memLen, vc16, Lpad);
      }
      // scores = (Q @ Kc^T) / 8  -> softmax -> probs @ Vc
      launch_gemm(stream, q16, DM, DH, kT16, Lpad, (long)DH * Lpad,
                  nullptr, 0, 0, s16, Lpad, (long)HL * Lpad,
                  nullptr, nullptr, 0, HL, Lpad, DH, 0.125f, 0, NH);
      k_softmax<<<dim3(HL, NH), dim3(256), 0, stream>>>(s16, HL, L, Lpad);
      launch_gemm(stream, s16, Lpad, (long)HL * Lpad, vc16, DH, (long)Lpad * DH,
                  nullptr, 0, 0, y16, DM, DH,
                  nullptr, nullptr, 0, HL, DH, Lpad, 1.f, 0, NH);
      // output proj + residual
      launch_gemm(stream, y16, DM, 0, Wo16[l], DM, 0, h32, DM, 0, nullptr, 0, 0,
                  nullptr, h32, DM, HL, DM, DM, 1.f, 0, 1);
      // append this chunk's SST k/v to memory (used from next chunk on)
      k_update_mem<<<dim3((GROUPS * DM + 255) / 256), dim3(256), 0, stream>>>(
          k16, v16, mk, mv, memLen);
      // FFN + residual
      k_rms<<<dim3(HL), dim3(256), 0, stream>>>(h32, n2g[l], hn16, DM);
      launch_gemm(stream, hn16, DM, 0, W116[l], DFF, 0, nullptr, 0, 0, ff16, DFF, 0,
                  nullptr, nullptr, 0, HL, DFF, DM, 1.f, 1, 1);
      launch_gemm(stream, ff16, DFF, 0, W216[l], DM, 0, h32, DM, 0, nullptr, 0, 0,
                  nullptr, h32, DM, HL, DM, DFF, 1.f, 0, 1);
    }
    k_slide<<<dim3((GROUPS * DM + 255) / 256), dim3(256), 0, stream>>>(h32, smem16, SSTR * c);
  }

  // ---- report projection: q = gelu(report@Wt1+b1)@Wt2+b2 ----
  launch_gemm(stream, rep16, D_T, 0, Wt1, DM, 0, nullptr, 0, 0, t2_16, DM, 0,
              tp_b1, nullptr, 0, NREP, DM, D_T, 1.f, 1, 1);
  launch_gemm(stream, t2_16, DM, 0, Wt2, DM, 0, rq32, DM, 0, nullptr, 0, 0,
              tp_b2, nullptr, 0, NREP, DM, DM, 1.f, 0, 1);

  // ---- cross attention over 256 SST memory tokens ----
  k_rms<<<dim3(NREP), dim3(256), 0, stream>>>(rq32, rn1, rqn16, DM);
  launch_gemm(stream, rqn16, DM, 0, Wcq, DM, 0, nullptr, 0, 0, cq16, DM, 0,
              nullptr, nullptr, 0, NREP, DM, DM, 1.f, 0, 1);
  launch_gemm(stream, smem16, DM, 0, Wck, DM, 0, nullptr, 0, 0, ck16, DM, 0,
              nullptr, nullptr, 0, MEMMAX, DM, DM, 1.f, 0, 1);
  launch_gemm(stream, smem16, DM, 0, Wcv, DM, 0, nullptr, 0, 0, cv16, DM, 0,
              nullptr, nullptr, 0, MEMMAX, DM, DM, 1.f, 0, 1);
  {
    long tot = (long)NH * DH * MEMMAX;
    k_build_kT<<<dim3((unsigned)((tot + 255) / 256)), dim3(256), 0, stream>>>(
        ck16, MEMMAX, nullptr, 0, kT16, MEMMAX);
    k_build_vc<<<dim3((unsigned)((tot + 255) / 256)), dim3(256), 0, stream>>>(
        cv16, MEMMAX, nullptr, 0, vc16, MEMMAX);
  }
  launch_gemm(stream, cq16, DM, DH, kT16, MEMMAX, (long)DH * MEMMAX,
              nullptr, 0, 0, s16, MEMMAX, (long)NREP * MEMMAX,
              nullptr, nullptr, 0, NREP, MEMMAX, DH, 0.125f, 0, NH);
  k_softmax<<<dim3(NREP, NH), dim3(256), 0, stream>>>(s16, NREP, MEMMAX, MEMMAX);
  launch_gemm(stream, s16, MEMMAX, (long)NREP * MEMMAX, vc16, DH, (long)MEMMAX * DH,
              nullptr, 0, 0, cy16, DM, DH,
              nullptr, nullptr, 0, NREP, DH, MEMMAX, 1.f, 0, NH);
  launch_gemm(stream, cy16, DM, 0, Wco, DM, 0, rq32, DM, 0, nullptr, 0, 0,
              nullptr, rq32, DM, NREP, DM, DM, 1.f, 0, 1);
  // ---- read-out FFN + head ----
  k_rms<<<dim3(NREP), dim3(256), 0, stream>>>(rq32, rn2, rqn16, DM);
  launch_gemm(stream, rqn16, DM, 0, Wrf1, DFF, 0, nullptr, 0, 0, cf1, DFF, 0,
              nullptr, nullptr, 0, NREP, DFF, DM, 1.f, 1, 1);
  launch_gemm(stream, cf1, DFF, 0, Wrf2, DM, 0, rq32, DM, 0, nullptr, 0, 0,
              nullptr, rq32, DM, NREP, DM, DFF, 1.f, 0, 1);
  k_head<<<dim3(1), dim3(768), 0, stream>>>(rq32, head_w, head_b, (float*)d_out);
}
